// NewCombinedLoss_49022756716805
// MI455X (gfx1250) — matrix operand order repaired and checked by reading, not compiled, not run on hardware
//
#include <hip/hip_runtime.h>
#include <hip/hip_fp16.h>

// ---------------- problem constants (match reference) ----------------
#define DHW    262144        // 64*64*64
#define NB     4
#define NC     4
#define NPAIR  12            // B * (C-1) boundary masks
#define INF_F  1.0e12f
#define T_ALPHA 0.3f
#define T_BETA  0.7f
#define SMOOTH  1e-5f

typedef __attribute__((ext_vector_type(16))) _Float16 v16h;
typedef __attribute__((ext_vector_type(8)))  float    v8f;

__device__ __forceinline__ float f4c(const float4& v, int e) {
  // e is always a compile-time constant after unrolling
  return (e == 0) ? v.x : (e == 1) ? v.y : (e == 2) ? v.z : v.w;
}

// Workspace float layout:
//  [0..15]   TP[b*4+c]
//  [16..31]  SUMP[b*4+c]
//  [32..47]  CNT[b*4+c]
//  [48]      focal sum
//  [49..60]  contrib[pair]  (pair = b*3 + (c-1)), signed out-in accumulation
//  [64..]    W field: 12 * DHW floats (squared-distance working buffer)

__global__ void zero_acc_kernel(float* __restrict__ acc) {
  if (threadIdx.x < 64) acc[threadIdx.x] = 0.f;
}

// ------------- fused softmax + focal + tversky statistics -------------
__global__ void stats_kernel(const float* __restrict__ preds,
                             const long long* __restrict__ targets,
                             float* __restrict__ acc) {
  __shared__ float sh[13];
  const int tid = threadIdx.x;
  if (tid < 13) sh[tid] = 0.f;
  __syncthreads();

  const int idx = blockIdx.x * 256 + tid;        // grid exactly covers NB*DHW
  const int b = idx >> 18;                       // DHW == 2^18
  const int v = idx & (DHW - 1);

  const float* q = preds + (size_t)b * NC * DHW + v;
  __builtin_prefetch(q + 1024, 0, 1);            // global_prefetch_b8 (gfx1250)

  const float x0 = q[0], x1 = q[DHW], x2 = q[2 * DHW], x3 = q[3 * DHW];
  const float mx = fmaxf(fmaxf(x0, x1), fmaxf(x2, x3));
  const float e0 = __expf(x0 - mx), e1 = __expf(x1 - mx);
  const float e2 = __expf(x2 - mx), e3 = __expf(x3 - mx);
  const float s  = e0 + e1 + e2 + e3;
  const float inv = __builtin_amdgcn_rcpf(s);    // s in [1,4]: well-conditioned
  float pv[4] = { e0 * inv, e1 * inv, e2 * inv, e3 * inv };

  const int t = (int)targets[(size_t)b * DHW + v];
  const float xt = (t == 0) ? x0 : (t == 1) ? x1 : (t == 2) ? x2 : x3;
  const float ce = -(xt - mx - __logf(s));
  const float pt = (t == 0) ? pv[0] : (t == 1) ? pv[1] : (t == 2) ? pv[2] : pv[3];
  const float om = 1.0f - pt;

  float vals[13];
  #pragma unroll
  for (int c = 0; c < 4; ++c) {
    const bool hit = (t == c);
    vals[c]     = pv[c];               // SUMP
    vals[4 + c] = hit ? pv[c] : 0.f;   // TP
    vals[8 + c] = hit ? 1.f : 0.f;     // CNT
  }
  vals[12] = om * om * ce;             // focal term

  #pragma unroll
  for (int k = 0; k < 13; ++k)
    for (int off = 16; off > 0; off >>= 1)
      vals[k] += __shfl_down(vals[k], off, 32);

  if ((tid & 31) == 0) {
    #pragma unroll
    for (int k = 0; k < 13; ++k) atomicAdd(&sh[k], vals[k]);
  }
  __syncthreads();
  if (tid < 13) {
    int slot;
    if (tid < 4)       slot = 16 + b * 4 + tid;        // SUMP
    else if (tid < 8)  slot = b * 4 + (tid - 4);       // TP
    else if (tid < 12) slot = 32 + b * 4 + (tid - 8);  // CNT
    else               slot = 48;                      // focal
    atomicAdd(&acc[slot], sh[tid]);
  }
}

// ------------------------- EDT field init ----------------------------
// seedIsMask==1 : f = 0 where (t==c) else INF   (-> dist_out^2)
// seedIsMask==0 : f = 0 where (t!=c) else INF   (-> dist_in^2)
__global__ void edt_init_kernel(const long long* __restrict__ targets,
                                float* __restrict__ W, int seedIsMask) {
  const int idx = blockIdx.x * 256 + threadIdx.x;    // covers NPAIR*DHW exactly
  const int m = idx >> 18;
  const int v = idx & (DHW - 1);
  const int b = m / 3;
  const int c = m - b * 3 + 1;
  const int t = (int)targets[(size_t)b * DHW + v];
  const int in = (t == c) ? 1 : 0;
  W[idx] = (in == seedIsMask) ? 0.f : INF_F;
}

// ---------------- separable min-plus squared DT, one axis ------------
// 64-long lines staged in LDS; out[i] = min_j f[j] + (i-j)^2, in place.
__global__ void edt_pass_kernel(float* __restrict__ W, int axis) {
  __shared__ float line[4][64];
  const int tid = threadIdx.x;
  const int li  = tid >> 6;            // line within block (4 lines/block)
  const int i   = tid & 63;
  const int gl  = blockIdx.x * 4 + li; // global line id, < NPAIR*4096
  const int m   = gl >> 12;
  const int r   = gl & 4095;

  size_t base; int stride;
  if (axis == 0)      { base = (size_t)m * DHW + (size_t)r * 64;                 stride = 1;    }
  else if (axis == 1) { base = (size_t)m * DHW + (size_t)(r >> 6) * 4096 + (r & 63); stride = 64; }
  else                { base = (size_t)m * DHW + r;                              stride = 4096; }

  line[li][i] = W[base + (size_t)i * stride];
  __syncthreads();

  const float fi = (float)i;
  float best = 3.0e30f;
  #pragma unroll 8
  for (int j = 0; j < 64; ++j) {
    const float d = fi - (float)j;
    best = fminf(best, __fmaf_rn(d, d, line[li][j]));
  }
  W[base + (size_t)i * stride] = best;
}

// -------- boundary contraction: diag( sqrt(W) x probs^T ) via WMMA ---
// A row m = pair (sqrt of squared distance), B col n = pair prob p_{c}(v).
// 16x16x32 f16 WMMA, f32 accumulation chained across K-chunks of 32.
// Pad rows/cols >=12 are zeroed via a 0/1 scale (no EXEC divergence).
__global__ void contrib_wmma_kernel(const float* __restrict__ W,
                                    const float* __restrict__ preds,
                                    float* __restrict__ contrib, float sign) {
  const int lane   = threadIdx.x & 31;
  const bool hiH   = lane >= 16;
  const int waveId = (blockIdx.x * blockDim.x + threadIdx.x) >> 5;
  const int nwaves = (gridDim.x * blockDim.x) >> 5;

  const int mA      = lane & 15;                       // A-matrix row (pair)
  const bool mValid = (mA < NPAIR);
  const int mRow    = mValid ? mA : 0;                 // clamp: read row 0, scale 0
  const float aScl  = mValid ? 1.f : 0.f;

  const int nN      = lane & 15;                       // B-matrix col (pair)
  const bool nValid = (nN < NPAIR);
  const int bB      = nValid ? (nN / 3) : 0;
  const int cB      = nValid ? (nN - (nN / 3) * 3 + 1) : 1;
  const float bScl  = nValid ? 1.f : 0.f;

  const int kbaseA = hiH ? 16 : 0;   // A K-group : kbaseA+[0..15]
  const int kbaseB = hiH ? 16 : 0;   // B K-group : kbaseB+[0..15]

  v8f acc = {};
  const int CHUNKS = DHW / 32;

  for (int ch = waveId; ch < CHUNKS; ch += nwaves) {
    const int vbase = ch * 32;

    // ---- A operand: 4x float4 (two pairs of 8 consecutive floats) ----
    const float4* wrow = (const float4*)(W + (size_t)mRow * DHW + vbase);
    const int g0 = kbaseA >> 2;                        // 0 or 4
    float4 wv[4];
    wv[0] = wrow[g0];     wv[1] = wrow[g0 + 1];        // K = kbaseA + 0..7
    wv[2] = wrow[g0 + 2]; wv[3] = wrow[g0 + 3];        // K = kbaseA + 8..15

    v16h a;
    #pragma unroll
    for (int g = 0; g < 4; ++g)
      #pragma unroll
      for (int e = 0; e < 4; ++e)
        a[g * 4 + e] = (_Float16)(aScl * __builtin_amdgcn_sqrtf(f4c(wv[g], e)));

    // ---- B operand: 16 consecutive voxels, 4 channels, float4 loads ----
    const float* qb = preds + (size_t)bB * NC * DHW + vbase + kbaseB;
    float4 X0[4], X1[4], X2[4], X3[4];
    #pragma unroll
    for (int g = 0; g < 4; ++g) {
      X0[g] = *(const float4*)(qb + 0 * DHW + g * 4);
      X1[g] = *(const float4*)(qb + 1 * DHW + g * 4);
      X2[g] = *(const float4*)(qb + 2 * DHW + g * 4);
      X3[g] = *(const float4*)(qb + 3 * DHW + g * 4);
    }

    v16h bm;
    #pragma unroll
    for (int g = 0; g < 4; ++g) {
      #pragma unroll
      for (int e = 0; e < 4; ++e) {
        const float x0 = f4c(X0[g], e), x1 = f4c(X1[g], e);
        const float x2 = f4c(X2[g], e), x3 = f4c(X3[g], e);
        const float mx = fmaxf(fmaxf(x0, x1), fmaxf(x2, x3));
        const float e0 = __expf(x0 - mx), e1 = __expf(x1 - mx);
        const float e2 = __expf(x2 - mx), e3 = __expf(x3 - mx);
        const float ec = (cB == 1) ? e1 : (cB == 2) ? e2 : e3;
        const float rs = __builtin_amdgcn_rcpf(e0 + e1 + e2 + e3);  // v_rcp_f32
        bm[g * 4 + e] = (_Float16)(bScl * ec * rs);
      }
    }

    acc = __builtin_amdgcn_wmma_f32_16x16x32_f16(
        /*neg_a=*/false, a, /*neg_b=*/false, bm,
        /*c_mod=*/(short)0, acc, /*reuse_a=*/false, /*reuse_b=*/false);
  }

  // diagonal extract: C/D layout VGPR r -> M = r + (hi?8:0), N = lane&15
  #pragma unroll
  for (int r = 0; r < 8; ++r) {
    const int M = r + (hiH ? 8 : 0);
    if (M == (lane & 15) && M < NPAIR)
      atomicAdd(&contrib[M], sign * acc[r]);
  }
}

// --------------------------- final scalar ----------------------------
__global__ void finalize_kernel(const float* __restrict__ acc,
                                float* __restrict__ out) {
  float tvsum = 0.f;
  for (int bc = 0; bc < 16; ++bc) {
    const float tp = acc[bc], sp = acc[16 + bc], cnt = acc[32 + bc];
    const float fp = sp - tp, fn = cnt - tp;
    tvsum += (tp + SMOOTH) / (tp + T_ALPHA * fp + T_BETA * fn + SMOOTH);
  }
  const float l_dice = 1.f - tvsum * (1.f / 16.f);
  const float l_main = acc[48] / (float)(NB * DHW);
  float bsum = 0.f;
  for (int m = 0; m < NPAIR; ++m) {
    const int b = m / 3, c = m % 3 + 1;
    if (acc[32 + b * 4 + c] > 0.f)          // nonempty gate
      bsum += acc[49 + m] / (float)DHW;     // mean over spatial dims
  }
  const float l_bound = bsum / ((float)(NB * 3) + 1e-8f);
  out[0] = l_dice + l_main + 0.01f * l_bound;
}

extern "C" void kernel_launch(void* const* d_in, const int* in_sizes, int n_in,
                              void* d_out, int out_size, void* d_ws, size_t ws_size,
                              hipStream_t stream) {
  (void)in_sizes; (void)n_in; (void)out_size; (void)ws_size;
  const float*     preds   = (const float*)d_in[0];
  const long long* targets = (const long long*)d_in[1];
  float* acc = (float*)d_ws;
  float* W   = acc + 64;
  float* out = (float*)d_out;

  zero_acc_kernel<<<1, 64, 0, stream>>>(acc);
  stats_kernel<<<(NB * DHW) / 256, 256, 0, stream>>>(preds, targets, acc);

  for (int phase = 0; phase < 2; ++phase) {
    edt_init_kernel<<<(NPAIR * DHW) / 256, 256, 0, stream>>>(targets, W,
                                                             phase == 0 ? 1 : 0);
    for (int axis = 0; axis < 3; ++axis)
      edt_pass_kernel<<<(NPAIR * 4096) / 4, 256, 0, stream>>>(W, axis);
    contrib_wmma_kernel<<<256, 256, 0, stream>>>(W, preds, acc + 49,
                                                 phase == 0 ? 1.0f : -1.0f);
  }
  finalize_kernel<<<1, 1, 0, stream>>>(acc, out);
}
